// PredictionHead_41162966565495
// MI455X (gfx1250) — matrix operand order repaired and checked by reading, not compiled, not run on hardware
//
#include <hip/hip_runtime.h>
#include <hip/hip_bf16.h>

// ---------------------------------------------------------------------------
// Problem constants (from reference)
// ---------------------------------------------------------------------------
#define BB   4
#define HH   512
#define WW   512
#define FF   256          // feature dim (GEMM K)
#define QQ   4096         // queries
#define QD   256          // query dim
#define HD   1024         // hidden dim (GEMM N)
#define WS   5            // window
#define NPOS 25           // WS*WS
#define MROWS 32          // 25 padded to 32 (two 16-row M tiles)
#define KPADF 260         // f32 LDS row: 256 + 4 dword pad (TDM pad_amount)
#define KPADH 264         // bf16 LDS row: 256 + 8 half pad (528B, 16B aligned)

typedef __attribute__((ext_vector_type(16))) __bf16 v16bf;
typedef __attribute__((ext_vector_type(8)))  __bf16 v8bf;
typedef __attribute__((ext_vector_type(8)))  float  v8f;
typedef __attribute__((ext_vector_type(4)))  unsigned int u32x4;
typedef __attribute__((ext_vector_type(8)))  unsigned int u32x8;

__device__ __forceinline__ unsigned short f2bf_bits(float x) {
    union { float f; unsigned u; } a; a.f = x;
    unsigned r = a.u + 0x7FFFu + ((a.u >> 16) & 1u);   // round-to-nearest-even
    return (unsigned short)(r >> 16);
}
__device__ __forceinline__ float bf2f(unsigned short s) {
    union { unsigned u; float f; } o; o.u = ((unsigned)s) << 16;
    return o.f;
}
__device__ __forceinline__ unsigned rfl(unsigned x) {
    return (unsigned)__builtin_amdgcn_readfirstlane((int)x);
}

// A-matrix fragment (16-bit, 16x32 MxK): lane L holds row M=L%16; halves
// e=0..7 are K=8g..8g+7, e=8..15 are K=16+8g..23+8g (g=L/16).
__device__ __forceinline__ v16bf load_a_frag_bf(const __bf16* row, int kb, int g) {
    v8bf lo = *(const v8bf*)(row + kb + 8 * g);
    v8bf hi = *(const v8bf*)(row + kb + 16 + 8 * g);
    v16bf a;
#pragma unroll
    for (int i = 0; i < 8; ++i) { a[i] = lo[i]; a[8 + i] = hi[i]; }
    return a;
}

// B-matrix fragment (16-bit, 32x16 KxN): lane L holds column N=L%16,
// halves e=0..7 are K=8g..8g+7, e=8..15 are K=16+8g..23+8g (g=L/16).
__device__ __forceinline__ v16bf load_b_frag(const __bf16* row, int kb, int g) {
    v8bf lo = *(const v8bf*)(row + kb + 8 * g);
    v8bf hi = *(const v8bf*)(row + kb + 16 + 8 * g);
    v16bf b;
#pragma unroll
    for (int i = 0; i < 8; ++i) { b[i] = lo[i]; b[8 + i] = hi[i]; }
    return b;
}

// ---------------------------------------------------------------------------
// Pre-pass kernels
// ---------------------------------------------------------------------------
__global__ void cvt_bf16_kernel(const float* __restrict__ in,
                                __bf16* __restrict__ out, int n) {
    int i = blockIdx.x * blockDim.x + threadIdx.x;
    if (i < n) out[i] = (__bf16)in[i];
}

// w1 is (F, HD) row-major; produce W1t[h][f] (K-contiguous for B fragments)
__global__ void w1_transpose_kernel(const float* __restrict__ w1,
                                    __bf16* __restrict__ w1t) {
    int i = blockIdx.x * blockDim.x + threadIdx.x;
    if (i < FF * HD) {
        int f = i >> 10;
        int h = i & (HD - 1);
        w1t[h * FF + f] = (__bf16)w1[i];
    }
}

// c_q = dot(queries[q], b2); one wave per query
__global__ void cq_kernel(const float* __restrict__ queries,
                          const float* __restrict__ b2,
                          float* __restrict__ cq) {
    int q    = blockIdx.x * 8 + (threadIdx.x >> 5);
    int lane = threadIdx.x & 31;
    float s = 0.f;
    for (int i = lane; i < QD; i += 32) s += queries[(size_t)q * QD + i] * b2[i];
#pragma unroll
    for (int m = 16; m >= 1; m >>= 1) s += __shfl_xor(s, m, 32);
    if (lane == 0) cq[q] = s;
}

// U = queries @ w2^T  (M=4096, K=256, N=1024), bf16 in / bf16 out, f32 acc.
__global__ __launch_bounds__(256)
void u_gemm_kernel(const __bf16* __restrict__ Qbf,
                   const __bf16* __restrict__ W2bf,
                   __bf16* __restrict__ U) {
    const int wave = threadIdx.x >> 5;
    const int lane = threadIdx.x & 31;
    const int g    = lane >> 4;
    const int ln   = lane & 15;

    const int mb = blockIdx.x * 32;
    const int nb = blockIdx.y * 128 + wave * 16;

    const __bf16* arow0 = Qbf + (size_t)(mb + ln) * QD;
    const __bf16* arow1 = arow0 + 16 * QD;
    const __bf16* brow  = W2bf + (size_t)(nb + ln) * QD;

    v8f acc0 = {}; v8f acc1 = {};
#pragma unroll
    for (int ks = 0; ks < 8; ++ks) {
        const int kb = ks * 32;
        v16bf bf = load_b_frag(brow, kb, g);
        v16bf a0 = load_a_frag_bf(arow0, kb, g);
        v16bf a1 = load_a_frag_bf(arow1, kb, g);
        acc0 = __builtin_amdgcn_wmma_f32_16x16x32_bf16(false, a0, false, bf,
                                                       (short)0, acc0, false, false);
        acc1 = __builtin_amdgcn_wmma_f32_16x16x32_bf16(false, a1, false, bf,
                                                       (short)0, acc1, false, false);
    }
    const int n = nb + ln;
#pragma unroll
    for (int r = 0; r < 8; ++r) {
        int m0 = mb + r + 8 * g;
        U[(size_t)m0 * HD + n]        = (__bf16)acc0[r];
        U[(size_t)(m0 + 16) * HD + n] = (__bf16)acc1[r];
    }
}

// ---------------------------------------------------------------------------
// Main kernel: one block per query, 8 wave32 waves.
// Keys staged by the Tensor Data Mover (gather mode, 32-bit row indices,
// LDS row padding 256->260 dwords for bank staggering), then converted once
// to a bf16 LDS tile with native v_cvt (pk) converts.
// logits[m] = relu(key_m . W1 + b1) . u_q + c_q   (-inf where padded)
// ---------------------------------------------------------------------------
__global__ __launch_bounds__(256)
void mask_head_kernel(const float* __restrict__ feat,        // (B,H,W,F)
                      const __bf16* __restrict__ W1t,        // (HD,F) bf16
                      const float* __restrict__ b1,          // (HD)
                      const __bf16* __restrict__ U,          // (Q,HD) bf16
                      const float* __restrict__ Cq,          // (Q)
                      const int* __restrict__ query_batch,
                      const int* __restrict__ query_i,
                      const int* __restrict__ query_j,
                      float* __restrict__ outLogit,          // (Q,25)
                      int* __restrict__ outIdx) {            // (Q,25,4)
    __shared__ float  sKeysF[MROWS * KPADF];        // 33280 B, f32 keys (TDM dest)
    __shared__ __bf16 sKeysBF[MROWS * KPADH];       // 16896 B, bf16 tile
    __shared__ __bf16 sU[HD];                       // u_q, bf16
    __shared__ float  sB1[HD];
    __shared__ float  sLogit[MROWS];
    __shared__ int    sIC[NPOS], sJC[NPOS], sValid[NPOS];
    __shared__ int    sPBI;
    __shared__ float  sCq;

    const int tid  = threadIdx.x;
    const int wave = tid >> 5;
    const int lane = tid & 31;
    const int g    = lane >> 4;
    const int ln   = lane & 15;
    const int q    = blockIdx.x;

    const int b  = query_batch[q];
    const int qi = query_i[q];
    const int qj = query_j[q];

    // --- Issue TDM gather of the 25 key rows (wave 0 only; EXEC ignored) ---
    // Every runtime-derived descriptor word is forced uniform (readfirstlane)
    // so the "s" asm constraints can be honored (VIMAGE needs SGPR groups).
    if (wave == 0) {
        const int sb  = (int)rfl((unsigned)b);
        const int sqi = (int)rfl((unsigned)qi);
        const int sqj = (int)rfl((unsigned)qj);
        const unsigned long long gaddr =
            (unsigned long long)(size_t)(feat + (size_t)sb * HH * WW * FF);
        const unsigned glo = rfl((unsigned)gaddr);
        const unsigned ghi = rfl((unsigned)(gaddr >> 32));
        const unsigned ldsbase = rfl((unsigned)(size_t)(&sKeysF[0]));
        auto rowIndex = [&](int k) -> unsigned {
            int oy = k / WS - WS / 2, ox = k % WS - WS / 2;
            int ii = min(max(sqi + oy, 0), HH - 1);
            int jj = min(max(sqj + ox, 0), WW - 1);
            return rfl((unsigned)(ii * WW + jj));
        };
#pragma unroll
        for (int t = 0; t < 4; ++t) {
            const unsigned nr = (t < 3) ? 8u : 1u;   // 8+8+8+1 = 25 rows
            u32x4 g0 = {}; u32x8 g1 = {}; u32x4 g2 = {}; u32x4 g3 = {};
            // Group 0: valid, gather_mode, 32-bit indices, lds/global addr, type=2
            g0[0] = 1u | (1u << 30) | (1u << 31);
            g0[1] = ldsbase + (unsigned)(t * 8 * KPADF * 4);
            g0[2] = glo;
            g0[3] = (ghi & 0x01FFFFFFu) | (2u << 30);
            // Group 1: data_size=4B, pad_enable, pad_interval=256dw, pad_amount=4dw
            g1[0] = (2u << 16) | (1u << 20) | (7u << 22) | (3u << 25);
            g1[1] = (FF & 0xFFFFu) << 16;                     // tensor_dim0 lo16
            g1[2] = (FF >> 16) | (((HH * WW) & 0xFFFFu) << 16);
            g1[3] = ((HH * WW) >> 16) | ((unsigned)FF << 16); // tensor_dim1 hi | tile_dim0
            g1[4] = nr;                                       // tile_dim1 = #indices
            g1[5] = (unsigned)FF;                             // tensor_dim0_stride lo
            g1[6] = 0u;
            g1[7] = 0u;
            const int base = t * 8;
#pragma unroll
            for (int j = 0; j < 4; ++j)
                g2[j] = (base + j < NPOS) ? rowIndex(base + j) : 0u;
#pragma unroll
            for (int j = 0; j < 4; ++j)
                g3[j] = (base + 4 + j < NPOS) ? rowIndex(base + 4 + j) : 0u;
            asm volatile("tensor_load_to_lds %0, %1, %2, %3"
                         :: "s"(g0), "s"(g1), "s"(g2), "s"(g3)
                         : "memory");
        }
    }

    // --- Overlapped with the TDM: per-position metadata + u_q/b1 staging ---
    if (tid < NPOS) {
        int oy = tid / WS - WS / 2, ox = tid % WS - WS / 2;
        int ii = qi + oy, jj = qj + ox;
        int v  = (ii >= 0) & (ii < HH) & (jj >= 0) & (jj < WW);
        sIC[tid]    = min(max(ii, 0), HH - 1);
        sJC[tid]    = min(max(jj, 0), WW - 1);
        sValid[tid] = v;
    }
    if (tid == 32) {   // lower_bound(query_batch, b) -> per-batch index
        int lo = 0, hi = QQ;
        while (lo < hi) { int mid = (lo + hi) >> 1;
                          if (query_batch[mid] < b) lo = mid + 1; else hi = mid; }
        sPBI = q - lo;
        sCq  = Cq[q];
    }
    if (tid < MROWS) sLogit[tid] = 0.f;
    for (int i = tid; i < HD; i += 256) {
        sU[i]  = U[(size_t)q * HD + i];
        sB1[i] = b1[i];
    }

    if (wave == 0) __builtin_amdgcn_s_wait_tensorcnt(0);
    __syncthreads();

    // --- One cooperative f32 -> bf16 conversion pass (native v_cvt) ---
    // Rows 25..31 hold garbage; their M-rows are independent and masked later.
#pragma unroll
    for (int m = wave; m < MROWS; m += 8) {
        const float* src = sKeysF + m * KPADF + lane * 8;
        v8bf d;
#pragma unroll
        for (int i = 0; i < 8; ++i) d[i] = (__bf16)src[i];
        *(v8bf*)(sKeysBF + m * KPADH + lane * 8) = d;
    }
    __syncthreads();

    // --- Hoist all 16 A fragments (keys are N-chunk invariant) ---
    v16bf afrag[16];
#pragma unroll
    for (int mt = 0; mt < 2; ++mt) {
        const __bf16* row = sKeysBF + (mt * 16 + ln) * KPADH;
#pragma unroll
        for (int ks = 0; ks < 8; ++ks)
            afrag[mt * 8 + ks] = load_a_frag_bf(row, ks * 32, g);
    }

    float lacc[16];
#pragma unroll
    for (int r = 0; r < 16; ++r) lacc[r] = 0.f;

    // 64 N-chunks of 16 hidden columns; wave w owns chunks w, w+8, ...
    for (int c = wave; c < HD / 16; c += 8) {
        const int nb = c * 16;
        const __bf16* w1row = W1t + (size_t)(nb + ln) * FF;

        v8f acc0 = {}; v8f acc1 = {};
#pragma unroll
        for (int ks = 0; ks < 8; ++ks) {
            v16bf bf = load_b_frag(w1row, ks * 32, g);
            acc0 = __builtin_amdgcn_wmma_f32_16x16x32_bf16(false, afrag[ks], false, bf,
                                                           (short)0, acc0, false, false);
            acc1 = __builtin_amdgcn_wmma_f32_16x16x32_bf16(false, afrag[8 + ks], false, bf,
                                                           (short)0, acc1, false, false);
        }
        // Fuse bias + ReLU + dot with u_q. Lane holds N = nb+ln, M = r+8g (+16).
        const int   n  = nb + ln;
        const float bb = sB1[n];
        const float uu = (float)sU[n];
#pragma unroll
        for (int r = 0; r < 8; ++r) {
            float h0 = acc0[r] + bb; h0 = h0 > 0.f ? h0 : 0.f;
            float h1 = acc1[r] + bb; h1 = h1 > 0.f ? h1 : 0.f;
            lacc[r]     += h0 * uu;
            lacc[8 + r] += h1 * uu;
        }
    }

    // Reduce over the 16 lanes sharing each M, then combine waves via LDS.
#pragma unroll
    for (int r = 0; r < 16; ++r) {
        float v = lacc[r];
#pragma unroll
        for (int m = 8; m >= 1; m >>= 1) v += __shfl_xor(v, m, 16);
        lacc[r] = v;
    }
    if (ln == 0) {
        const int mOff = 8 * g;   // lane0 -> M=r,16+r ; lane16 -> M=8+r,24+r
#pragma unroll
        for (int r = 0; r < 8; ++r) {
            atomicAdd(&sLogit[mOff + r],      lacc[r]);
            atomicAdd(&sLogit[16 + mOff + r], lacc[8 + r]);
        }
    }
    __syncthreads();

    if (tid < NPOS) {
        float val = sLogit[tid] + sCq;
        if (!sValid[tid]) val = -__builtin_huge_valf();
        outLogit[(size_t)q * NPOS + tid] = val;
        int* oi = outIdx + ((size_t)q * NPOS + tid) * 4;
        oi[0] = b; oi[1] = sIC[tid]; oi[2] = sJC[tid]; oi[3] = sPBI;
    }
}

// ---------------------------------------------------------------------------
// Launch
// ---------------------------------------------------------------------------
extern "C" void kernel_launch(void* const* d_in, const int* in_sizes, int n_in,
                              void* d_out, int out_size, void* d_ws, size_t ws_size,
                              hipStream_t stream) {
    const float* queries = (const float*)d_in[0];
    const float* feat    = (const float*)d_in[1];
    const float* w1      = (const float*)d_in[2];
    const float* b1      = (const float*)d_in[3];
    const float* w2      = (const float*)d_in[4];
    const float* b2      = (const float*)d_in[5];
    const int*   qb      = (const int*)d_in[6];
    const int*   qi      = (const int*)d_in[7];
    const int*   qj      = (const int*)d_in[8];

    // Workspace layout (bytes)
    char* ws = (char*)d_ws;
    __bf16* Qbf  = (__bf16*)(ws);                    // 2 MB
    __bf16* W1t  = (__bf16*)(ws + 2097152);          // 512 KB
    __bf16* W2bf = (__bf16*)(ws + 2621440);          // 512 KB
    __bf16* U    = (__bf16*)(ws + 3145728);          // 8 MB
    float*  Cq   = (float*)(ws + 11534336);          // 16 KB

    float* outLogit = (float*)d_out;                 // Q*25 f32
    int*   outIdx   = (int*)d_out + QQ * NPOS;       // Q*25*4 i32

    cvt_bf16_kernel<<<(QQ * QD + 255) / 256, 256, 0, stream>>>(queries, Qbf, QQ * QD);
    cvt_bf16_kernel<<<(HD * QD + 255) / 256, 256, 0, stream>>>(w2, W2bf, HD * QD);
    w1_transpose_kernel<<<(FF * HD + 255) / 256, 256, 0, stream>>>(w1, W1t);
    cq_kernel<<<QQ / 8, 256, 0, stream>>>(queries, b2, Cq);
    u_gemm_kernel<<<dim3(QQ / 32, HD / 128), 256, 0, stream>>>(Qbf, W2bf, U);
    mask_head_kernel<<<QQ, 256, 0, stream>>>(feat, W1t, b1, U, Cq,
                                             qb, qi, qj, outLogit, outIdx);
}
